// MultiHeadAttention_29738353558235
// MI455X (gfx1250) — compile-verified, matrix operand order checked by reading, not run on hardware
//
#include <hip/hip_runtime.h>
#include <hip/hip_bf16.h>

typedef __bf16 bf16_t;
typedef __attribute__((ext_vector_type(16))) __bf16 v16bf;
typedef __attribute__((ext_vector_type(8)))  __bf16 v8bf;
typedef __attribute__((ext_vector_type(4)))  __bf16 v4bf;
typedef __attribute__((ext_vector_type(8)))  float  v8f;

// D = A(16x32 bf16) * B(32x16 bf16) + C(16x16 f32), wave32.
__device__ __forceinline__ v8f wmma_bf16(v16bf a, v16bf b, v8f c) {
  return __builtin_amdgcn_wmma_f32_16x16x32_bf16(
      /*neg_a=*/false, a, /*neg_b=*/false, b,
      /*c_mod=*/(short)0, c, /*reuse_a=*/false, /*reuse_b=*/false);
}

// CDNA5 async memory->LDS copy (16B per active lane), tracked by ASYNCcnt.
// lds_addr: this lane's LDS byte address; voff: 32-bit byte offset added to
// the uniform 64-bit SGPR base (GVS addressing mode).
__device__ __forceinline__ void async_load_b128(unsigned lds_addr,
                                                const void* sbase,
                                                unsigned voff) {
  asm volatile("global_load_async_to_lds_b128 %0, %1, %2"
               :: "v"(lds_addr), "v"(voff),
                  "s"((unsigned long long)(uintptr_t)sbase)
               : "memory");
}
__device__ __forceinline__ void wait_async0() {
  asm volatile("s_wait_asynccnt 0" ::: "memory");
}

// Load one 16x32 bf16 WMMA fragment (A layout; B uses the mirrored layout with
// lane = column) from LDS. Per ISA 7.12.2: lanes 0-15 hold K {0..7,16..23},
// lanes 16-31 hold K {8..15,24..31} -> two 16-byte ds loads per lane.
__device__ __forceinline__ v16bf frag16(const bf16_t* rowp, int lane) {
  const int half = lane >> 4;
  v8bf lo = *(const v8bf*)(rowp + half * 8);
  v8bf hi = *(const v8bf*)(rowp + 16 + half * 8);
  v16bf f;
#pragma unroll
  for (int i = 0; i < 8; ++i) { f[i] = lo[i]; f[i + 8] = hi[i]; }
  return f;
}

// C[M,N] = A[M,K] @ W[K,N] + bias.  Block tile 128x128, 8 waves as 2x4,
// each wave owns a 64x32 sub-tile (4x2 grid of 16x16 WMMA tiles).
template <bool A_BF16, bool OUT_BF16>
__global__ __launch_bounds__(256) void gemm_bias_kernel(
    const void* __restrict__ Av, const float* __restrict__ W,
    const float* __restrict__ bias, void* __restrict__ Cv,
    int M, int N, int K) {
  __shared__ bf16_t sA[128 * 32];  // [m][k]
  __shared__ bf16_t sB[128 * 32];  // transposed: [n][k]

  const int tid  = threadIdx.x;
  const int lane = tid & 31;
  const int wave = tid >> 5;
  const int wm   = wave & 1;   // 0..1 -> 64 rows each
  const int wn   = wave >> 1;  // 0..3 -> 32 cols each
  const int half = lane >> 4;
  const int nl   = lane & 15;
  const int bm   = blockIdx.x;
  const int bn   = blockIdx.y;

  v8f acc[4][2] = {};

  for (int k0 = 0; k0 < K; k0 += 32) {
    __syncthreads();
    // ---- stage A tile (128 x 32) into LDS as bf16 ----
    if constexpr (A_BF16) {
      // already bf16: pure byte copy -> async DMA to LDS (no VGPR roundtrip)
      const bf16_t* A = (const bf16_t*)Av;
      const unsigned sA_lds = (unsigned)(uintptr_t)(void*)sA;
#pragma unroll
      for (int i = tid; i < 512; i += 256) {  // 512 x 16B chunks
        int r = i >> 2, c = (i & 3) * 8;
        async_load_b128(sA_lds + (unsigned)(r * 32 + c) * 2, A,
                        (unsigned)((bm * 128 + r) * K + k0 + c) * 2);
      }
    } else {
      const float* A = (const float*)Av;
#pragma unroll
      for (int i = tid * 4; i < 128 * 32; i += 1024) {
        int m = i >> 5, k = i & 31;
        float4 f = *(const float4*)&A[(size_t)(bm * 128 + m) * K + k0 + k];
        v4bf bv;
        bv[0] = (bf16_t)f.x; bv[1] = (bf16_t)f.y;
        bv[2] = (bf16_t)f.z; bv[3] = (bf16_t)f.w;
        *(v4bf*)&sA[m * 32 + k] = bv;
      }
    }
    // ---- stage B tile (32 x 128) transposed into [n][k], fp32->bf16 ----
#pragma unroll
    for (int i = tid; i < 32 * 128; i += 256) {
      int k = i >> 7, n = i & 127;
      sB[n * 32 + k] = (bf16_t)W[(size_t)(k0 + k) * N + bn * 128 + n];
    }
    if constexpr (A_BF16) wait_async0();
    __syncthreads();

    v16bf af[4], bfr[2];
#pragma unroll
    for (int tm = 0; tm < 4; ++tm)
      af[tm] = frag16(&sA[(wm * 64 + tm * 16 + nl) * 32], lane);
#pragma unroll
    for (int tn = 0; tn < 2; ++tn)
      bfr[tn] = frag16(&sB[(wn * 32 + tn * 16 + nl) * 32], lane);
#pragma unroll
    for (int tm = 0; tm < 4; ++tm)
#pragma unroll
      for (int tn = 0; tn < 2; ++tn)
        acc[tm][tn] = wmma_bf16(af[tm], bfr[tn], acc[tm][tn]);
  }

  // epilogue: C layout -> row m = tile + half*8 + r, col n = tile + nl
#pragma unroll
  for (int tm = 0; tm < 4; ++tm)
#pragma unroll
    for (int tn = 0; tn < 2; ++tn)
#pragma unroll
      for (int r = 0; r < 8; ++r) {
        int m = bm * 128 + wm * 64 + tm * 16 + half * 8 + r;
        int n = bn * 128 + wn * 32 + tn * 16 + nl;
        float v = acc[tm][tn][r] + bias[n];
        if (OUT_BF16) ((bf16_t*)Cv)[(size_t)m * N + n] = (bf16_t)v;
        else          ((float*)Cv)[(size_t)m * N + n] = v;
      }
}

// Flash attention, causal, one block per (b, h, 128-query block).
// 8 waves; each wave owns 16 query rows. K/V processed in 64-key blocks.
// Q and K tiles are DMA'd to LDS with async loads (already bf16); V goes
// through VGPRs because it is stored transposed [dh][key].
__global__ __launch_bounds__(256) void flash_attn_kernel(
    const bf16_t* __restrict__ qkv, bf16_t* __restrict__ out) {
  constexpr int S = 2048, D = 1024, Dh = 64;
  __shared__ bf16_t sQ[128 * 64];      // [q][dh]
  __shared__ bf16_t sK[64 * 64];       // [key][dh]   (contraction over dh)
  __shared__ bf16_t sVt[64 * 64];      // [dh][key]   (contraction over key)
  __shared__ bf16_t sP[8 * 16 * 64];   // per-wave P tile [16 q][64 key]

  const int tid = threadIdx.x, lane = tid & 31, wave = tid >> 5;
  const int half = lane >> 4, nl = lane & 15;
  const int qb = blockIdx.x, h = blockIdx.y, b = blockIdx.z;

  const bf16_t* qbase = qkv + (size_t)b * S * 3 * D + (size_t)h * Dh;
  const bf16_t* kbase = qbase + D;
  const bf16_t* vbase = qbase + 2 * D;

  const unsigned sQ_lds = (unsigned)(uintptr_t)(void*)sQ;
  const unsigned sK_lds = (unsigned)(uintptr_t)(void*)sK;

  // async-stage Q tile: 128x64 bf16 = 1024 x 16B chunks, 4 per thread
#pragma unroll
  for (int i = tid; i < 1024; i += 256) {
    int r = i >> 3, c = (i & 7) * 8;
    async_load_b128(sQ_lds + (unsigned)(r * 64 + c) * 2, qbase,
                    (unsigned)((qb * 128 + r) * 3 * D + c) * 2);
  }

  float row_max[8], row_sum[8];
#pragma unroll
  for (int r = 0; r < 8; ++r) { row_max[r] = -1e30f; row_sum[r] = 0.f; }
  v8f o[4] = {};

  const int nkb = qb * 2 + 2;  // causal: keys 0 .. qb*128+127
  bf16_t* pw = &sP[wave * 16 * 64];

  for (int j = 0; j < nkb; ++j) {
    __syncthreads();
    // async-stage K tile: 64x64 bf16 = 512 x 16B chunks, 2 per thread
#pragma unroll
    for (int i = tid; i < 512; i += 256) {
      int r = i >> 3, c = (i & 7) * 8;
      async_load_b128(sK_lds + (unsigned)(r * 64 + c) * 2, kbase,
                      (unsigned)((j * 64 + r) * 3 * D + c) * 2);
    }
    // V tile through VGPRs with transpose into [dh][key]
#pragma unroll
    for (int i = tid * 4; i < 64 * 64; i += 1024) {
      int r = i >> 6, c = i & 63;
      v4bf vv = *(const v4bf*)&vbase[(size_t)(j * 64 + r) * 3 * D + c];
#pragma unroll
      for (int t = 0; t < 4; ++t) sVt[(c + t) * 64 + r] = vv[t];
    }
    if (j + 1 < nkb)  // prefetch next V block into caches
      __builtin_prefetch(&vbase[(size_t)((j + 1) * 64 + (tid >> 2)) * 3 * D], 0, 1);
    wait_async0();
    __syncthreads();

    // scores S = Q(16x64) @ K^T(64x64): contraction over dh, 2 k-steps
    v8f s[4] = {};
#pragma unroll
    for (int ks = 0; ks < 2; ++ks) {
      v16bf aq = frag16(&sQ[(wave * 16 + nl) * 64 + ks * 32], lane);
#pragma unroll
      for (int tn = 0; tn < 4; ++tn) {
        v16bf bk = frag16(&sK[(tn * 16 + nl) * 64 + ks * 32], lane);
        s[tn] = wmma_bf16(aq, bk, s[tn]);
      }
    }

    // online softmax; row m = wave*16 + half*8 + r lives in 16 lanes of a half
#pragma unroll
    for (int r = 0; r < 8; ++r) {
      const int qrow = qb * 128 + wave * 16 + half * 8 + r;
      float mx = -1e30f;
#pragma unroll
      for (int tn = 0; tn < 4; ++tn) {
        int key = j * 64 + tn * 16 + nl;
        float sv = s[tn][r] * 0.125f;          // 1/sqrt(64)
        sv = (key <= qrow) ? sv : -1e30f;      // causal (diag included)
        s[tn][r] = sv;
        mx = fmaxf(mx, sv);
      }
#pragma unroll
      for (int off = 1; off < 16; off <<= 1)
        mx = fmaxf(mx, __shfl_xor(mx, off, 32));
      const float m_new = fmaxf(row_max[r], mx);
      const float alpha = __expf(row_max[r] - m_new);
      row_max[r] = m_new;
      float lsum = 0.f;
#pragma unroll
      for (int tn = 0; tn < 4; ++tn) {
        float p = __expf(s[tn][r] - m_new);
        s[tn][r] = p;
        lsum += p;
      }
#pragma unroll
      for (int off = 1; off < 16; off <<= 1)
        lsum += __shfl_xor(lsum, off, 32);
      row_sum[r] = row_sum[r] * alpha + lsum;
#pragma unroll
      for (int tn = 0; tn < 4; ++tn) o[tn][r] *= alpha;
    }

    // spill P (bf16) to this wave's private LDS; same-wave DS ordering
#pragma unroll
    for (int tn = 0; tn < 4; ++tn)
#pragma unroll
      for (int r = 0; r < 8; ++r)
        pw[(half * 8 + r) * 64 + tn * 16 + nl] = (bf16_t)s[tn][r];
    asm volatile("s_wait_dscnt 0" ::: "memory");

    // o += P(16x64) @ V(64x64): contraction over key, 2 k-steps
#pragma unroll
    for (int ks = 0; ks < 2; ++ks) {
      v16bf ap = frag16(&pw[nl * 64 + ks * 32], lane);
#pragma unroll
      for (int tn = 0; tn < 4; ++tn) {
        v16bf bv = frag16(&sVt[(tn * 16 + nl) * 64 + ks * 32], lane);
        o[tn] = wmma_bf16(ap, bv, o[tn]);
      }
    }
  }

  // normalize and write merged-head output (bf16) for the final projection
#pragma unroll
  for (int tn = 0; tn < 4; ++tn)
#pragma unroll
    for (int r = 0; r < 8; ++r) {
      int qrow = qb * 128 + wave * 16 + half * 8 + r;
      out[((size_t)b * S + qrow) * D + h * Dh + tn * 16 + nl] =
          (bf16_t)(o[tn][r] / row_sum[r]);
    }
}

extern "C" void kernel_launch(void* const* d_in, const int* in_sizes, int n_in,
                              void* d_out, int out_size, void* d_ws, size_t ws_size,
                              hipStream_t stream) {
  (void)in_sizes; (void)n_in; (void)out_size; (void)ws_size;
  const float* x      = (const float*)d_in[0];
  const float* w_qkv  = (const float*)d_in[1];
  const float* b_qkv  = (const float*)d_in[2];
  const float* w_proj = (const float*)d_in[3];
  const float* b_proj = (const float*)d_in[4];
  float* out = (float*)d_out;

  const int B = 4, S = 2048, D = 1024;
  const int M = B * S;  // 8192

  bf16_t* qkv  = (bf16_t*)d_ws;             // [M, 3D] bf16 = 48 MB
  bf16_t* attn = qkv + (size_t)M * 3 * D;   // [M, D]  bf16 = 16 MB

  dim3 blk(256);

  // 1) QKV projection: fp32 in -> bf16 out
  dim3 g1(M / 128, (3 * D) / 128);
  gemm_bias_kernel<false, true><<<g1, blk, 0, stream>>>(
      x, w_qkv, b_qkv, qkv, M, 3 * D, D);

  // 2) causal flash attention per (b, h, q-block)
  dim3 g2(S / 128, D / 64, B);  // (16, 16, 4)
  flash_attn_kernel<<<g2, blk, 0, stream>>>(qkv, attn);

  // 3) output projection: bf16 in -> fp32 out
  dim3 g3(M / 128, D / 128);
  gemm_bias_kernel<true, false><<<g3, blk, 0, stream>>>(
      attn, w_proj, b_proj, out, M, D, D);
}
